// Head_37623913513539
// MI455X (gfx1250) — matrix operand order repaired and checked by reading, not compiled, not run on hardware
//
#include <hip/hip_runtime.h>
#include <hip/hip_bf16.h>

// ---------------------------------------------------------------------------
// Fused latent attention (MLA-style) for MI455X / gfx1250, wave32 + WMMA bf16.
//   q   = (x @ Wd_w^T + b) * 1/sqrt(64)
//   S   = q @ Wdkv^T  (causal)
//   out = softmax(S) @ Wdkv       (flash/online softmax, never materialize S)
// Grid: (T/32, B).  Block: 128 threads = 4 waves; each wave owns one 16x16
// score tile and 9 16x16 output tiles ([32 x 288] outputs per block).
// ---------------------------------------------------------------------------

typedef __attribute__((ext_vector_type(4)))  float  v4f;
typedef __attribute__((ext_vector_type(8)))  float  v8f;
typedef __attribute__((ext_vector_type(4)))  __bf16 v4bf;
typedef __attribute__((ext_vector_type(8)))  __bf16 v8bf;
typedef __attribute__((ext_vector_type(16))) __bf16 v16bf;

#define BB    4
#define TT    2048
#define CC    1024
#define LL    288
#define BM    32      // query rows per block (also key-chunk size)
#define QSTR  296     // bf16 row stride for q_s / kv_s (288 + 8 pad, 592B rows)
#define SSTR  36      // f32 row stride for score tile
#define PSTR  40      // bf16 row stride for P tile
#define X1STR 72      // bf16 row stride for x chunk (64 + 8 pad)

__device__ __forceinline__ v8f wmma_bf16(v16bf a, v16bf b, v8f c) {
  // D = A(16x32 bf16) x B(32x16 bf16) + C(16x16 f32)
  return __builtin_amdgcn_wmma_f32_16x16x32_bf16(
      false, a, false, b, (short)0, c, false, false);
}

// Build a 16-element bf16 fragment from two contiguous 8-element LDS chunks.
__device__ __forceinline__ v16bf ld_frag(const __bf16* p, int lo, int hi) {
  v8bf a = *(const v8bf*)(p + lo);
  v8bf b = *(const v8bf*)(p + hi);
  return __builtin_shufflevector(a, b, 0,1,2,3,4,5,6,7,8,9,10,11,12,13,14,15);
}

__global__ __launch_bounds__(128) void mla_attn_kernel(
    const float* __restrict__ x,     // [B, T, C]
    const float* __restrict__ Wdkv,  // [B, T, L]
    const float* __restrict__ Wd_w,  // [L, C]
    const float* __restrict__ Wd_b,  // [L]
    float* __restrict__ out)         // [B, T, L]
{
  __shared__ __attribute__((aligned(16))) __bf16 q_s[BM * QSTR];       // 18944 B
  __shared__ __attribute__((aligned(16))) union {
    struct {                                   // phase 1: q projection staging
      __bf16 xs[BM * X1STR];                   // x chunk   [32 x 64]   4608 B
      __bf16 ws[LL * 64];                      // Wd_w chunk[288 x 64] 36864 B
    } p1;
    struct {                                   // phase 2/3: attention staging
      __bf16 kv [BM * QSTR];                   // Wdkv chunk row-major  [32 x 288]
      __bf16 kvT[LL * 32];                     // Wdkv chunk transposed [288 x 32]
      float  sco[BM * SSTR];                   // raw scores (f32)
      __bf16 p  [BM * PSTR];                   // probabilities (bf16)
    } p2;                                      // 44544 B
  } u;
  __shared__ float m_s[BM], l_s[BM], a_s[BM], red[BM * 4];

  const int tid  = threadIdx.x;
  const int lane = tid & 31;
  const int wid  = tid >> 5;
  const int mt   = wid & 1;       // which 16-row half of the 32 queries
  const int ng   = wid >> 1;      // N-group (score col half / output tile group)
  const int hl   = lane >> 4;     // lane half (WMMA 16-bit layout selector)
  const int ln   = lane & 15;
  const int t0   = blockIdx.x * BM;
  const int b    = blockIdx.y;

  const float* xb  = x    + ((size_t)b * TT + t0) * CC;
  const float* kvb = Wdkv + (size_t)b * TT * LL;

  const int arow = mt * 16 + ln;  // A-fragment row for this lane
  const int ao8  = hl * 16;       // A-fragment K offset (contiguous k order, k = hl*16 + s)

  // ---------------- Phase 1: q = (x @ Wd_w^T + bias) * 1/8 ----------------
  v8f qacc[9];
  #pragma unroll
  for (int j = 0; j < 9; ++j) qacc[j] = v8f{};

  for (int kc = 0; kc < CC; kc += 64) {
    // Stage x chunk [32 x 64] as bf16 (float4 loads, 8B bf16 stores).
    #pragma unroll
    for (int e4 = tid; e4 < BM * 16; e4 += 128) {
      int r = e4 >> 4, c = (e4 & 15) * 4;
      v4f v = *(const v4f*)&xb[(size_t)r * CC + kc + c];
      *(v4bf*)&u.p1.xs[r * X1STR + c] = __builtin_convertvector(v, v4bf);
    }
    // Stage Wd_w chunk [288 x 64] as bf16.
    for (int e4 = tid; e4 < LL * 16; e4 += 128) {
      int r = e4 >> 4, c = (e4 & 15) * 4;
      v4f v = *(const v4f*)&Wd_w[(size_t)r * CC + kc + c];
      *(v4bf*)&u.p1.ws[r * 64 + c] = __builtin_convertvector(v, v4bf);
    }
    __syncthreads();
    #pragma unroll
    for (int k0 = 0; k0 < 64; k0 += 32) {
      v16bf af = ld_frag(u.p1.xs, arow * X1STR + k0 + ao8,
                                  arow * X1STR + k0 + ao8 + 8);
      int bc0 = (ng * 9) * 16 + ln;
      v16bf bf = ld_frag(u.p1.ws, bc0 * 64 + k0 + hl * 16,
                                  bc0 * 64 + k0 + hl * 16 + 8);
      #pragma unroll
      for (int j = 0; j < 9; ++j) {          // software-pipelined B loads
        v16bf bnext = bf;
        if (j < 8) {
          int bc = (ng * 9 + j + 1) * 16 + ln;
          bnext = ld_frag(u.p1.ws, bc * 64 + k0 + hl * 16,
                                   bc * 64 + k0 + hl * 16 + 8);
        }
        qacc[j] = wmma_bf16(af, bf, qacc[j]);
        bf = bnext;
      }
    }
    __syncthreads();
  }
  // C layout: VGPR i -> row (mt*16 + hl*8 + i), col fixed per lane.
  #pragma unroll
  for (int j = 0; j < 9; ++j) {
    int ncol = (ng * 9 + j) * 16 + ln;
    float bias = Wd_b[ncol];
    #pragma unroll
    for (int i = 0; i < 8; ++i) {
      int row = mt * 16 + hl * 8 + i;
      q_s[row * QSTR + ncol] = (__bf16)((qacc[j][i] + bias) * 0.125f);
    }
  }
  if (tid < BM) { m_s[tid] = -__builtin_inff(); l_s[tid] = 0.f; }
  __syncthreads();

  // ---------------- Phase 2/3: flash attention over key chunks -------------
  v8f oacc[9];
  #pragma unroll
  for (int j = 0; j < 9; ++j) oacc[j] = v8f{};

  const int rrow = tid & 31;     // softmax: row handled by this thread
  const int rg   = tid >> 5;     // softmax: 8-column group

  for (int s0 = 0; s0 <= t0; s0 += BM) {
    // Stage the Wdkv chunk: row-major (scores B) + transposed (output B).
    for (int e4 = tid; e4 < BM * 72; e4 += 128) {      // 18 float4 per thread
      int r = e4 / 72;
      int c = (e4 - r * 72) * 4;
      v4f v = *(const v4f*)&kvb[(size_t)(s0 + r) * LL + c];
      v4bf h = __builtin_convertvector(v, v4bf);
      *(v4bf*)&u.p2.kv[r * QSTR + c] = h;
      u.p2.kvT[(c + 0) * 32 + r] = h.x;
      u.p2.kvT[(c + 1) * 32 + r] = h.y;
      u.p2.kvT[(c + 2) * 32 + r] = h.z;
      u.p2.kvT[(c + 3) * 32 + r] = h.w;
    }
    // Prefetch next chunk into cache while this one is consumed.
    if (s0 + BM <= t0) {
      int pr = tid >> 2, pc = (tid & 3) * 72;
      __builtin_prefetch(&kvb[(size_t)(s0 + BM + pr) * LL + pc], 0, 1);
    }
    __syncthreads();

    // Scores: this wave computes tile (mt, ng) of the 32x32 score block.
    v8f sacc = v8f{};
    {
      const int bn = ng * 16 + ln;                     // key row (N index)
      v16bf af = ld_frag(q_s, arow * QSTR + ao8, arow * QSTR + ao8 + 8);
      v16bf bf = ld_frag(u.p2.kv, bn * QSTR + hl * 16,
                                  bn * QSTR + hl * 16 + 8);
      #pragma unroll
      for (int k0 = 0; k0 < LL; k0 += 32) {            // pipelined A/B loads
        v16bf anext = af, bnext = bf;
        if (k0 + 32 < LL) {
          anext = ld_frag(q_s, arow * QSTR + k0 + 32 + ao8,
                               arow * QSTR + k0 + 32 + ao8 + 8);
          bnext = ld_frag(u.p2.kv, bn * QSTR + k0 + 32 + hl * 16,
                                   bn * QSTR + k0 + 32 + hl * 16 + 8);
        }
        sacc = wmma_bf16(af, bf, sacc);
        af = anext; bf = bnext;
      }
    }
    #pragma unroll
    for (int i = 0; i < 8; ++i) {
      int row  = mt * 16 + hl * 8 + i;
      int scol = ng * 16 + ln;
      float v = sacc[i];
      if (s0 + scol > t0 + row) v = -__builtin_inff(); // causal mask
      u.p2.sco[row * SSTR + scol] = v;
    }
    __syncthreads();

    // Online softmax (all 128 threads; 4 threads per row).
    float pm = -__builtin_inff();
    #pragma unroll
    for (int c = 0; c < 8; ++c)
      pm = fmaxf(pm, u.p2.sco[rrow * SSTR + rg * 8 + c]);
    red[rrow * 4 + rg] = pm;
    __syncthreads();
    if (tid < BM) {
      float mo = m_s[tid];
      float mn = fmaxf(fmaxf(red[tid*4+0], red[tid*4+1]),
                       fmaxf(red[tid*4+2], red[tid*4+3]));
      mn = fmaxf(mo, mn);
      m_s[tid] = mn;
      a_s[tid] = __expf(mo - mn);                      // alpha (0 on 1st chunk)
    }
    __syncthreads();
    {
      float mn = m_s[rrow];
      float ps = 0.f;
      #pragma unroll
      for (int c = 0; c < 8; ++c) {
        int col = rg * 8 + c;
        float p = __expf(u.p2.sco[rrow * SSTR + col] - mn);  // exp(-inf)=0
        ps += p;
        u.p2.p[rrow * PSTR + col] = (__bf16)p;
      }
      red[rrow * 4 + rg] = ps;
    }
    __syncthreads();
    if (tid < BM)
      l_s[tid] = l_s[tid] * a_s[tid] +
                 (red[tid*4+0] + red[tid*4+1] + red[tid*4+2] + red[tid*4+3]);
    __syncthreads();

    // Rescale running output by alpha (per-row), then O += P @ Wdkv_chunk.
    float al[8];
    #pragma unroll
    for (int i = 0; i < 8; ++i) al[i] = a_s[mt * 16 + hl * 8 + i];
    #pragma unroll
    for (int j = 0; j < 9; ++j) {
      #pragma unroll
      for (int i = 0; i < 8; ++i) oacc[j][i] *= al[i];
    }
    {
      v16bf af = ld_frag(u.p2.p, arow * PSTR + ao8, arow * PSTR + ao8 + 8);
      int bl0 = (ng * 9) * 16 + ln;
      v16bf bf = ld_frag(u.p2.kvT, bl0 * 32 + hl * 16, bl0 * 32 + hl * 16 + 8);
      #pragma unroll
      for (int j = 0; j < 9; ++j) {          // software-pipelined B loads
        v16bf bnext = bf;
        if (j < 8) {
          int bl = (ng * 9 + j + 1) * 16 + ln;
          bnext = ld_frag(u.p2.kvT, bl * 32 + hl * 16, bl * 32 + hl * 16 + 8);
        }
        oacc[j] = wmma_bf16(af, bf, oacc[j]);
        bf = bnext;
      }
    }
    __syncthreads();
  }

  // ---------------- Finalize: out = O / l ----------------------------------
  if (tid < BM) a_s[tid] = 1.0f / l_s[tid];
  __syncthreads();
  float inv[8];
  #pragma unroll
  for (int i = 0; i < 8; ++i) inv[i] = a_s[mt * 16 + hl * 8 + i];
  float* ob = out + ((size_t)b * TT + t0) * LL;
  #pragma unroll
  for (int j = 0; j < 9; ++j) {
    int ncol = (ng * 9 + j) * 16 + ln;
    #pragma unroll
    for (int i = 0; i < 8; ++i) {
      int row = mt * 16 + hl * 8 + i;
      ob[(size_t)row * LL + ncol] = oacc[j][i] * inv[i];
    }
  }
}

extern "C" void kernel_launch(void* const* d_in, const int* in_sizes, int n_in,
                              void* d_out, int out_size, void* d_ws, size_t ws_size,
                              hipStream_t stream) {
  const float* x    = (const float*)d_in[0];   // [4, 2048, 1024]
  const float* Wdkv = (const float*)d_in[1];   // [4, 2048, 288]
  const float* Wd_w = (const float*)d_in[2];   // [288, 1024]
  const float* Wd_b = (const float*)d_in[3];   // [288]
  float* out = (float*)d_out;                  // [4, 2048, 288]
  (void)in_sizes; (void)n_in; (void)out_size; (void)d_ws; (void)ws_size;

  dim3 grid(TT / BM, BB);
  dim3 block(128);
  hipLaunchKernelGGL(mla_attn_kernel, grid, block, 0, stream,
                     x, Wdkv, Wd_w, Wd_b, out);
}
